// SpectralEntropyAttention_74637941670112
// MI455X (gfx1250) — compile-verified, matrix operand order checked by reading, not run on hardware
//
#include <hip/hip_runtime.h>

// ---------------------------------------------------------------------------
// SpectralEntropyAttention for MI455X (gfx1250, wave32, WMMA).
// Heavy matmuls: v_wmma_f32_16x16x32_bf16 with fp32 accumulation, LDS
// double-buffered via global_load_async_to_lds_b128 (ASYNCcnt pipeline).
// Requires ~1.55 GB of workspace in d_ws.
// ---------------------------------------------------------------------------

typedef __attribute__((ext_vector_type(16))) __bf16 v16bf;
typedef __attribute__((ext_vector_type(8)))  float  v8f;

namespace {
constexpr int kB  = 2;
constexpr int kS  = 2048;
constexpr int kH  = 1024;
constexpr int kNH = 16;
constexpr int kHD = 64;      // kH / kNH
constexpr int kSD = 256;
constexpr int kBS = kB * kS;   // 4096 token rows
constexpr int kBH = kB * kNH;  // 32 head slices
constexpr float kEpsNorm = 1e-12f;
constexpr float kEpsLn   = 1e-5f;
} // namespace

__device__ __forceinline__ __bf16 f2bf(float f) { return (__bf16)f; }

// ---- CDNA5 async global->LDS copy (16 bytes per lane) ----------------------
__device__ __forceinline__ void async_ld16(unsigned lds_off, const void* gptr) {
    asm volatile("global_load_async_to_lds_b128 %0, %1, off"
                 :: "v"(lds_off), "v"(gptr)
                 : "memory");
}
__device__ __forceinline__ void wait_async0() {
#if __has_builtin(__builtin_amdgcn_s_wait_asynccnt)
    __builtin_amdgcn_s_wait_asynccnt(0);
#else
    asm volatile("s_wait_asynccnt 0x0" ::: "memory");
#endif
}

// ------------------------------ block reductions ---------------------------

__device__ __forceinline__ float blockReduceSum(float v) {
    __shared__ float red[256];
    red[threadIdx.x] = v;
    __syncthreads();
    for (int s = 128; s > 0; s >>= 1) {
        if ((int)threadIdx.x < s) red[threadIdx.x] += red[threadIdx.x + s];
        __syncthreads();
    }
    float r = red[0];
    __syncthreads();
    return r;
}

__device__ __forceinline__ float blockReduceMax(float v) {
    __shared__ float red[256];
    red[threadIdx.x] = v;
    __syncthreads();
    for (int s = 128; s > 0; s >>= 1) {
        if ((int)threadIdx.x < s) red[threadIdx.x] = fmaxf(red[threadIdx.x], red[threadIdx.x + s]);
        __syncthreads();
    }
    float r = red[0];
    __syncthreads();
    return r;
}

// ------------------------------ batched WMMA GEMM --------------------------
// C[bat, M, N] = scale * A[bat, M, K] * B[bat, N, K]^T (+ bias[N]) (+ add[bat, M, N])
// Optional secondary bf16 output C16 (same layout as C).
// A, B bf16 K-contiguous (lda = ldb = K); C fp32 (ldc = N).
// Block: 256 threads = 8 waves; each wave owns a 32x32 tile (2x2 WMMA frags).
// K staged in LDS in chunks of 32 with async double buffering.

template <int WM, int WN>
__launch_bounds__(256)
__global__ void gemm_bf16_kernel(const __bf16* __restrict__ A,
                                 const __bf16* __restrict__ B,
                                 float* __restrict__ C,
                                 __bf16* __restrict__ C16,
                                 int M, int N, int K,
                                 long long strideA, long long strideB, long long strideC,
                                 float scale,
                                 const float* __restrict__ bias,
                                 const float* __restrict__ add, long long strideAdd) {
    constexpr int BM = WM * 32;
    constexpr int BN = WN * 32;
    constexpr int KC = 32;

    __shared__ __align__(128) __bf16 As[2][BM][KC];
    __shared__ __align__(128) __bf16 Bs[2][BN][KC];

    const int tid   = threadIdx.x;
    const int lane  = tid & 31;
    const int wave  = tid >> 5;
    const int wm    = wave % WM;
    const int wn    = wave / WM;
    const int l16   = lane & 15;
    const int lhalf = lane >> 4;

    const int m0  = blockIdx.y * BM;
    const int n0  = blockIdx.x * BN;
    const int bat = blockIdx.z;

    const __bf16* Ab = A + (long long)bat * strideA;
    const __bf16* Bb = B + (long long)bat * strideB;
    float*        Cb = C + (long long)bat * strideC;

    const unsigned asBase = (unsigned)(uintptr_t)(&As[0][0][0]);
    const unsigned bsBase = (unsigned)(uintptr_t)(&Bs[0][0][0]);
    constexpr unsigned aBuf = (unsigned)(BM * KC * 2);
    constexpr unsigned bBuf = (unsigned)(BN * KC * 2);

    // Issue this thread's share of the async copies for one K-chunk.
    auto stage = [&](int buf, int kc) {
        #pragma unroll
        for (int c = tid; c < BM * 4; c += 256) {
            const int r  = c >> 2;
            const int cb = (c & 3) * 16;                 // byte offset within a 64B row
            async_ld16(asBase + (unsigned)buf * aBuf + (unsigned)(r * 64 + cb),
                       (const char*)(Ab + (long long)(m0 + r) * K + kc) + cb);
        }
        #pragma unroll
        for (int c = tid; c < BN * 4; c += 256) {
            const int r  = c >> 2;
            const int cb = (c & 3) * 16;
            async_ld16(bsBase + (unsigned)buf * bBuf + (unsigned)(r * 64 + cb),
                       (const char*)(Bb + (long long)(n0 + r) * K + kc) + cb);
        }
    };

    v8f acc[2][2] = {};

    stage(0, 0);
    const int nK = K / KC;
    for (int ik = 0; ik < nK; ++ik) {
        const int buf = ik & 1;
        wait_async0();       // this wave's copies into As/Bs[buf] have landed in LDS
        __syncthreads();     // everyone's have; prior-iteration reads of buf^1 are done
        if (ik + 1 < nK) stage(buf ^ 1, (ik + 1) * KC);

        // A fragment (16x32): lanes 0-15 -> K 0..15 of row M=lane, lanes 16-31 -> K 16..31.
        v16bf fa[2], fb[2];
        #pragma unroll
        for (int i = 0; i < 2; ++i)
            fa[i] = *reinterpret_cast<const v16bf*>(&As[buf][wm * 32 + i * 16 + l16][lhalf * 16]);
        #pragma unroll
        for (int j = 0; j < 2; ++j)
            fb[j] = *reinterpret_cast<const v16bf*>(&Bs[buf][wn * 32 + j * 16 + l16][lhalf * 16]);

        #pragma unroll
        for (int i = 0; i < 2; ++i)
            #pragma unroll
            for (int j = 0; j < 2; ++j)
                acc[i][j] = __builtin_amdgcn_wmma_f32_16x16x32_bf16(
                    false, fa[i], false, fb[j], (short)0, acc[i][j], false, false);
    }

    // Epilogue: VGPR r holds (row = 16*i + r + 8*(lane>>4), col = 16*j + lane&15).
    #pragma unroll
    for (int i = 0; i < 2; ++i) {
        #pragma unroll
        for (int j = 0; j < 2; ++j) {
            const int mBase = m0 + wm * 32 + i * 16 + lhalf * 8;
            const int n     = n0 + wn * 32 + j * 16 + l16;
            const float bv  = bias ? bias[n] : 0.0f;
            #pragma unroll
            for (int r = 0; r < 8; ++r) {
                const int m = mBase + r;
                float v = acc[i][j][r] * scale + bv;
                if (add) v += add[(long long)bat * strideAdd + (long long)m * N + n];
                Cb[(long long)m * N + n] = v;
                if (C16) C16[(long long)bat * strideC + (long long)m * N + n] = f2bf(v);
            }
        }
    }
}

// ------------------------------ elementwise kernels ------------------------

__global__ __launch_bounds__(256) void cvt_f32_bf16_kernel(const float* __restrict__ in,
                                                           __bf16* __restrict__ out, long n) {
    long i = (long)blockIdx.x * blockDim.x + threadIdx.x;
    if (i < n) out[i] = f2bf(in[i]);
}

// fp32 [B,S,H] -> bf16 [B*NH, S, HD]
__global__ __launch_bounds__(256) void split_heads_kernel(const float* __restrict__ in,
                                                          __bf16* __restrict__ out) {
    long i = (long)blockIdx.x * blockDim.x + threadIdx.x;
    if (i >= (long)kBS * kH) return;
    const int  d  = (int)(i % kHD);
    const int  h  = (int)((i / kHD) % kNH);
    const long bs = i / kH;
    const long b  = bs / kS;
    const long s  = bs % kS;
    out[((b * kNH + h) * (long)kS + s) * kHD + d] = f2bf(in[i]);
}

// fp32 [B,S,H] -> bf16 [B*NH, HD, S]  (transposed per head for p @ v)
__global__ __launch_bounds__(256) void heads_T_kernel(const float* __restrict__ in,
                                                      __bf16* __restrict__ out) {
    long i = (long)blockIdx.x * blockDim.x + threadIdx.x;
    if (i >= (long)kBS * kH) return;
    const int  d  = (int)(i % kHD);
    const int  h  = (int)((i / kHD) % kNH);
    const long bs = i / kH;
    const long b  = bs / kS;
    const long s  = bs % kS;
    out[((b * kNH + h) * (long)kHD + d) * kS + s] = f2bf(in[i]);
}

// fp32 [B*NH, S, HD] -> bf16 [B,S,H]
__global__ __launch_bounds__(256) void merge_heads_kernel(const float* __restrict__ in,
                                                          __bf16* __restrict__ out) {
    long i = (long)blockIdx.x * blockDim.x + threadIdx.x;
    if (i >= (long)kBS * kH) return;
    const int  d  = (int)(i % kHD);
    const int  h  = (int)((i / kHD) % kNH);
    const long bs = i / kH;
    const long b  = bs / kS;
    const long s  = bs % kS;
    out[i] = f2bf(in[((b * kNH + h) * (long)kS + s) * kHD + d]);
}

// batched fp32 [bat, S, D] -> bf16 [bat, D, S]
__global__ __launch_bounds__(256) void transpose_cvt_kernel(const float* __restrict__ in,
                                                            __bf16* __restrict__ out,
                                                            int S, int D) {
    const int  bat = blockIdx.z;
    long i = (long)blockIdx.x * blockDim.x + threadIdx.x;
    if (i >= (long)S * D) return;
    const int s = (int)(i / D);
    const int d = (int)(i % D);
    out[((long)bat * D + d) * S + s] = f2bf(in[((long)bat * S + s) * D + d]);
}

__global__ __launch_bounds__(256) void sigmoid2_cvt_kernel(const float* __restrict__ in,
                                                           __bf16* __restrict__ out, long n) {
    long i = (long)blockIdx.x * blockDim.x + threadIdx.x;
    if (i < n) {
        const float s = 1.0f / (1.0f + __expf(-in[i]));
        out[i] = f2bf(s * s);
    }
}

// attn[row] += w_d * patt[row] / max(||patt[row]||_2, eps); optionally also
// writes the refreshed bf16 copy consumed by the next fractal GEMM.
__global__ __launch_bounds__(256) void rownorm_update_kernel(const float* __restrict__ patt,
                                                             float* __restrict__ attn,
                                                             __bf16* __restrict__ attnb,
                                                             const float* __restrict__ fws,
                                                             int d, int S) {
    const long row = blockIdx.x;
    const float* pr = patt + row * (long)S;
    float*       ar = attn + row * (long)S;
    float ss = 0.0f;
    for (int i = threadIdx.x; i < S; i += 256) { const float v = pr[i]; ss += v * v; }
    ss = blockReduceSum(ss);
    const float inv = 1.0f / fmaxf(sqrtf(ss), kEpsNorm);
    const float w   = fws[d];
    for (int i = threadIdx.x; i < S; i += 256) {
        const float nv = ar[i] + w * (pr[i] * inv);
        ar[i] = nv;
        if (attnb) attnb[row * (long)S + i] = f2bf(nv);
    }
}

// row softmax, fp32 in -> bf16 out
__global__ __launch_bounds__(256) void softmax_cvt_kernel(const float* __restrict__ attn,
                                                          __bf16* __restrict__ pb, int S) {
    const long row = blockIdx.x;
    const float* ar = attn + row * (long)S;
    __bf16*      pr = pb + row * (long)S;
    float mx = -3.4e38f;
    for (int i = threadIdx.x; i < S; i += 256) mx = fmaxf(mx, ar[i]);
    mx = blockReduceMax(mx);
    float sum = 0.0f;
    for (int i = threadIdx.x; i < S; i += 256) sum += __expf(ar[i] - mx);
    sum = blockReduceSum(sum);
    const float inv = 1.0f / sum;
    for (int i = threadIdx.x; i < S; i += 256) pr[i] = f2bf(__expf(ar[i] - mx) * inv);
}

// out = LN(o + clip(fw)*sy + x) * g + b      (one block per token row, H = 1024)
__global__ __launch_bounds__(256) void fuse_ln_kernel(const float* __restrict__ o,
                                                      const float* __restrict__ sy,
                                                      const float* __restrict__ x,
                                                      const float* __restrict__ sfw,
                                                      const float* __restrict__ g,
                                                      const float* __restrict__ bta,
                                                      float* __restrict__ out) {
    const long row = blockIdx.x;
    const float fw = fminf(fmaxf(sfw[0], 0.0f), 0.5f);
    const float* orow = o  + row * (long)kH;
    const float* srow = sy + row * (long)kH;
    const float* xrow = x  + row * (long)kH;
    float z[4];
    float s1 = 0.0f, s2 = 0.0f;
    #pragma unroll
    for (int t = 0; t < 4; ++t) {
        const int i = threadIdx.x + t * 256;
        const float zz = orow[i] + fw * srow[i] + xrow[i];
        z[t] = zz; s1 += zz; s2 += zz * zz;
    }
    s1 = blockReduceSum(s1);
    s2 = blockReduceSum(s2);
    const float mu  = s1 * (1.0f / kH);
    const float var = s2 * (1.0f / kH) - mu * mu;
    const float inv = rsqrtf(var + kEpsLn);
    #pragma unroll
    for (int t = 0; t < 4; ++t) {
        const int i = threadIdx.x + t * 256;
        out[row * (long)kH + i] = (z[t] - mu) * inv * g[i] + bta[i];
    }
}

// ------------------------------ host orchestration -------------------------

extern "C" void kernel_launch(void* const* d_in, const int* in_sizes, int n_in,
                              void* d_out, int out_size, void* d_ws, size_t ws_size,
                              hipStream_t stream) {
    (void)in_sizes; (void)n_in; (void)out_size; (void)ws_size;

    const float* x    = (const float*)d_in[0];
    const float* mask = (const float*)d_in[1];
    const float* Wq   = (const float*)d_in[2];  const float* bq  = (const float*)d_in[3];
    const float* Wk   = (const float*)d_in[4];  const float* bk  = (const float*)d_in[5];
    const float* Wv   = (const float*)d_in[6];  const float* bv  = (const float*)d_in[7];
    const float* Wo   = (const float*)d_in[8];  const float* bo  = (const float*)d_in[9];
    const float* Wsq  = (const float*)d_in[10]; const float* bsq = (const float*)d_in[11];
    const float* Wsk  = (const float*)d_in[12]; const float* bsk = (const float*)d_in[13];
    const float* Wsv  = (const float*)d_in[14]; const float* bsv = (const float*)d_in[15];
    const float* Wst  = (const float*)d_in[16]; const float* bst = (const float*)d_in[17];
    const float* frw  = (const float*)d_in[18];
    const float* sfw  = (const float*)d_in[19];
    const float* lng  = (const float*)d_in[20];
    const float* lnb  = (const float*)d_in[21];
    float* out = (float*)d_out;

    char* wp = (char*)d_ws;
    auto carve = [&](size_t bytes) -> void* {
        void* r = (void*)wp;
        wp += (bytes + 255) & ~(size_t)255;
        return r;
    };
    const long SS   = (long)kS * kS;
    const long BHSS = (long)kBH * SS;

    __bf16* xb    = (__bf16*)carve((size_t)kBS * kH * 2);
    __bf16* Wqb   = (__bf16*)carve((size_t)kH * kH * 2);
    __bf16* Wkb   = (__bf16*)carve((size_t)kH * kH * 2);
    __bf16* Wvb   = (__bf16*)carve((size_t)kH * kH * 2);
    __bf16* Wob   = (__bf16*)carve((size_t)kH * kH * 2);
    __bf16* Wsqb  = (__bf16*)carve((size_t)kSD * kH * 2);
    __bf16* Wskb  = (__bf16*)carve((size_t)kSD * kH * 2);
    __bf16* Wsvb  = (__bf16*)carve((size_t)kSD * kH * 2);
    __bf16* Wstb  = (__bf16*)carve((size_t)kH * kSD * 2);
    float*  projf = (float*)carve((size_t)kBS * kH * 4);
    __bf16* qb    = (__bf16*)carve((size_t)kBH * kS * kHD * 2);
    __bf16* kb    = (__bf16*)carve((size_t)kBH * kS * kHD * 2);
    __bf16* vT    = (__bf16*)carve((size_t)kBH * kHD * kS * 2);
    float*  attn  = (float*)carve((size_t)BHSS * 4);
    __bf16* attnb = (__bf16*)carve((size_t)BHSS * 2);
    float*  patt  = (float*)carve((size_t)BHSS * 4);
    float*  yh    = (float*)carve((size_t)kBH * kS * kHD * 4);
    __bf16* yb    = (__bf16*)carve((size_t)kBS * kH * 2);
    float*  of    = (float*)carve((size_t)kBS * kH * 4);
    float*  qsf   = (float*)carve((size_t)kBS * kSD * 4);
    float*  ksf   = (float*)carve((size_t)kBS * kSD * 4);
    float*  vsf   = (float*)carve((size_t)kBS * kSD * 4);
    __bf16* qsb   = (__bf16*)carve((size_t)kBS * kSD * 2);
    __bf16* ksb   = (__bf16*)carve((size_t)kBS * kSD * 2);
    __bf16* vsT   = (__bf16*)carve((size_t)kB * kSD * kS * 2);
    float*  ssf   = (float*)carve((size_t)kB * SS * 4);
    __bf16* spb   = (__bf16*)carve((size_t)kB * SS * 2);
    float*  svf   = (float*)carve((size_t)kBS * kSD * 4);
    __bf16* svb   = (__bf16*)carve((size_t)kBS * kSD * 2);
    float*  syf   = (float*)carve((size_t)kBS * kH * 4);

    auto cvt = [&](const float* in, __bf16* o_, long n) {
        cvt_f32_bf16_kernel<<<dim3((unsigned)((n + 255) / 256)), 256, 0, stream>>>(in, o_, n);
    };
    auto gemm24 = [&](const __bf16* A, const __bf16* Bm, float* Cm, __bf16* C16,
                      int M, int N, int K,
                      long long sA, long long sB, long long sC, int batch, float scale,
                      const float* bias, const float* add, long long sAdd) {
        dim3 g((unsigned)(N / 128), (unsigned)(M / 64), (unsigned)batch);
        gemm_bf16_kernel<2, 4><<<g, 256, 0, stream>>>(A, Bm, Cm, C16, M, N, K, sA, sB, sC,
                                                      scale, bias, add, sAdd);
    };
    auto gemm42 = [&](const __bf16* A, const __bf16* Bm, float* Cm, __bf16* C16,
                      int M, int N, int K,
                      long long sA, long long sB, long long sC, int batch, float scale,
                      const float* bias, const float* add, long long sAdd) {
        dim3 g((unsigned)(N / 64), (unsigned)(M / 128), (unsigned)batch);
        gemm_bf16_kernel<4, 2><<<g, 256, 0, stream>>>(A, Bm, Cm, C16, M, N, K, sA, sB, sC,
                                                      scale, bias, add, sAdd);
    };

    const long nXH = (long)kBS * kH;
    const unsigned gXH = (unsigned)((nXH + 255) / 256);

    cvt(x,   xb,   nXH);
    cvt(Wq,  Wqb,  (long)kH * kH);
    cvt(Wk,  Wkb,  (long)kH * kH);
    cvt(Wv,  Wvb,  (long)kH * kH);
    cvt(Wo,  Wob,  (long)kH * kH);
    cvt(Wsq, Wsqb, (long)kSD * kH);
    cvt(Wsk, Wskb, (long)kSD * kH);
    cvt(Wsv, Wsvb, (long)kSD * kH);
    cvt(Wst, Wstb, (long)kH * kSD);

    gemm24(xb, Wqb, projf, nullptr, kBS, kH, kH, 0, 0, 0, 1, 1.0f, bq, nullptr, 0);
    split_heads_kernel<<<gXH, 256, 0, stream>>>(projf, qb);
    gemm24(xb, Wkb, projf, nullptr, kBS, kH, kH, 0, 0, 0, 1, 1.0f, bk, nullptr, 0);
    split_heads_kernel<<<gXH, 256, 0, stream>>>(projf, kb);
    gemm24(xb, Wvb, projf, nullptr, kBS, kH, kH, 0, 0, 0, 1, 1.0f, bv, nullptr, 0);
    heads_T_kernel<<<gXH, 256, 0, stream>>>(projf, vT);

    gemm24(qb, kb, attn, attnb, kS, kS, kHD,
           (long long)kS * kHD, (long long)kS * kHD, SS, kBH,
           0.125f, nullptr, mask, 0);

    for (int d = 0; d < 3; ++d) {
        gemm24(attnb, attnb, patt, nullptr, kS, kS, kS, SS, SS, SS, kBH,
               1.0f, nullptr, nullptr, 0);
        rownorm_update_kernel<<<(unsigned)(kBH * kS), 256, 0, stream>>>(
            patt, attn, (d < 2) ? attnb : nullptr, frw, d, kS);
    }

    softmax_cvt_kernel<<<(unsigned)(kBH * kS), 256, 0, stream>>>(attn, attnb, kS);
    gemm42(attnb, vT, yh, nullptr, kS, kHD, kS, SS, (long long)kHD * kS,
           (long long)kS * kHD, kBH, 1.0f, nullptr, nullptr, 0);
    merge_heads_kernel<<<gXH, 256, 0, stream>>>(yh, yb);

    gemm24(yb, Wob, of, nullptr, kBS, kH, kH, 0, 0, 0, 1, 1.0f, bo, nullptr, 0);

    gemm24(xb, Wsqb, qsf, qsb, kBS, kSD, kH, 0, 0, 0, 1, 1.0f, bsq, nullptr, 0);
    gemm24(xb, Wskb, ksf, ksb, kBS, kSD, kH, 0, 0, 0, 1, 1.0f, bsk, nullptr, 0);
    gemm24(xb, Wsvb, vsf, nullptr, kBS, kSD, kH, 0, 0, 0, 1, 1.0f, bsv, nullptr, 0);
    {
        dim3 g((unsigned)(((long)kS * kSD + 255) / 256), 1, (unsigned)kB);
        transpose_cvt_kernel<<<g, 256, 0, stream>>>(vsf, vsT, kS, kSD);
    }
    gemm24(qsb, ksb, ssf, nullptr, kS, kS, kSD,
           (long long)kS * kSD, (long long)kS * kSD, SS, kB,
           1.0f / 16.0f, nullptr, nullptr, 0);
    sigmoid2_cvt_kernel<<<(unsigned)(((long)kB * SS + 255) / 256), 256, 0, stream>>>(
        ssf, spb, (long)kB * SS);
    gemm24(spb, vsT, svf, svb, kS, kSD, kS, SS, (long long)kSD * kS,
           (long long)kS * kSD, kB, 1.0f, nullptr, nullptr, 0);
    gemm24(svb, Wstb, syf, nullptr, kBS, kH, kSD, 0, 0, 0, 1, 1.0f, bst, nullptr, 0);

    fuse_ln_kernel<<<(unsigned)kBS, 256, 0, stream>>>(of, syf, x, sfw, lng, lnb, out);
}
